// MultiHeadAttention_37288906064366
// MI455X (gfx1250) — matrix operand order repaired and checked by reading, not compiled, not run on hardware
//
#include <hip/hip_runtime.h>
#include <hip/hip_bf16.h>
#include <stdint.h>

typedef __bf16 bf16_t;
typedef __attribute__((ext_vector_type(16))) __bf16 v16bf;
typedef __attribute__((ext_vector_type(8)))  __bf16 v8bf;
typedef __attribute__((ext_vector_type(8)))  float  v8f;
typedef __attribute__((ext_vector_type(4)))  unsigned int u32x4;
typedef __attribute__((ext_vector_type(4)))  int i32x4;
typedef __attribute__((ext_vector_type(8)))  int i32x8;

static constexpr int B_   = 2;
static constexpr int T_   = 2048;
static constexpr int DIM_ = 2048;
static constexpr int NH_  = 16;
static constexpr int HD_  = 128;      // DIM/NH
static constexpr int M_   = B_ * T_;  // 4096

#if __has_builtin(__builtin_amdgcn_tensor_load_to_lds) && __has_builtin(__builtin_amdgcn_s_wait_tensorcnt)
#define USE_TDM 1
#else
#define USE_TDM 0
#endif

__device__ __forceinline__ bf16_t f2bf(float f) {
  uint32_t u = __builtin_bit_cast(uint32_t, f);
  uint32_t r = u + 0x7FFFu + ((u >> 16) & 1u);   // round-to-nearest-even
  uint16_t s = (uint16_t)(r >> 16);
  return __builtin_bit_cast(bf16_t, s);
}

// 16-lane-row reductions via DPP16 row_ror (pure VALU; keeps LDS pipe free).
#if __has_builtin(__builtin_amdgcn_update_dpp)
template<int CTRL>
__device__ __forceinline__ float dpp_rot(float v) {
  return __builtin_bit_cast(float,
      __builtin_amdgcn_update_dpp(0, __builtin_bit_cast(int, v), CTRL, 0xF, 0xF, true));
}
__device__ __forceinline__ float row_max16(float v) {
  v = fmaxf(v, dpp_rot<0x128>(v));  // row_ror:8
  v = fmaxf(v, dpp_rot<0x124>(v));  // row_ror:4
  v = fmaxf(v, dpp_rot<0x122>(v));  // row_ror:2
  v = fmaxf(v, dpp_rot<0x121>(v));  // row_ror:1
  return v;
}
__device__ __forceinline__ float row_sum16(float v) {
  v += dpp_rot<0x128>(v);
  v += dpp_rot<0x124>(v);
  v += dpp_rot<0x122>(v);
  v += dpp_rot<0x121>(v);
  return v;
}
#else
__device__ __forceinline__ float row_max16(float v) {
  for (int mk = 1; mk < 16; mk <<= 1) v = fmaxf(v, __shfl_xor(v, mk, 32));
  return v;
}
__device__ __forceinline__ float row_sum16(float v) {
  for (int mk = 1; mk < 16; mk <<= 1) v += __shfl_xor(v, mk, 32);
  return v;
}
#endif

#if USE_TDM
// Issue a TDM load of a 128-row x 32-col bf16 tile (row stride DIM_ elems)
// from global to LDS. One instruction per wave; tracked by TENSORcnt.
// D# per CDNA5 ISA ch.8: group0 = flags/lds_addr/global_addr/type,
// group1 = data_size + tensor dims + tile dims + strides, groups 2/3 = 0 (2D).
__device__ __forceinline__ void tdm_load_tile(bf16_t* lds_ptr, const bf16_t* gptr)
{
  uint64_t ga = (uint64_t)(uintptr_t)gptr;
  uint32_t lo = (uint32_t)(uintptr_t)lds_ptr;   // LDS byte offset (addr[31:0])
  u32x4 g0;
  g0[0] = 1u;                                                  // count=1 (valid)
  g0[1] = lo;                                                  // lds_addr
  g0[2] = (uint32_t)ga;                                        // global_addr[31:0]
  g0[3] = (uint32_t)((ga >> 32) & 0x01FFFFFFu) | (2u << 30);   // addr[56:32] | type=2
  i32x8 g1;
  g1[0] = (int)(1u << 16);                                     // data_size = 2 bytes
  g1[1] = (int)((2048u & 0xFFFFu) << 16);                      // tensor_dim0 lo16
  g1[2] = (int)(((2048u >> 16) & 0xFFFFu) | ((128u & 0xFFFFu) << 16)); // dim0 hi | dim1 lo
  g1[3] = (int)(32u << 16);                                    // dim1 hi=0 | tile_dim0=32
  g1[4] = (int)128u;                                           // tile_dim1=128, tile_dim2=0
  g1[5] = (int)2048u;                                          // tensor_dim0_stride lo32
  g1[6] = 0;                                                   // stride hi | dim1_stride lo
  g1[7] = 0;
  i32x4 gz = {0, 0, 0, 0};
#if __clang_major__ >= 23
  i32x8 gz8 = {0, 0, 0, 0, 0, 0, 0, 0};
  __builtin_amdgcn_tensor_load_to_lds(g0, g1, gz, gz, gz8, 0);
#else
  __builtin_amdgcn_tensor_load_to_lds(g0, g1, gz, gz, 0);
#endif
}
#endif

// ---------------------------------------------------------------------------
// Bulk f32 -> bf16 conversion (memory-bound, one-time).
// ---------------------------------------------------------------------------
__global__ __launch_bounds__(256)
void f32_to_bf16_kernel(const float* __restrict__ src, bf16_t* __restrict__ dst, int n)
{
  int i = (blockIdx.x * 256 + threadIdx.x) * 8;
  if (i >= n) return;
  v8bf o;
#pragma unroll
  for (int j = 0; j < 8; ++j) o[j] = f2bf(src[i + j]);
  *(v8bf*)(dst + i) = o;
}

// ---------------------------------------------------------------------------
// LDS-staged NT GEMM: C[m,n] = sum_k A[m,k] * W[n,k], bf16 in, f32 accum.
// Block tile 128x128, 8 waves as 4(M) x 2(N), each wave 32x64 (8 WMMA/K-step).
// TDM path: double-buffered LDS tiles, DMA for step i+1 overlaps WMMA of i.
// MODE 0: NeoX RoPE + transposed bf16 store; 1: transposed bf16; 2: f32.
// ---------------------------------------------------------------------------
template<int MODE>
__global__ __launch_bounds__(256)
void gemm_nt_kernel(const bf16_t* __restrict__ A, const bf16_t* __restrict__ W,
                    void* __restrict__ dstv)
{
#if USE_TDM
  __shared__ bf16_t a_lds[2][128 * 32];
  __shared__ bf16_t b_lds[2][128 * 32];
#else
  __shared__ bf16_t a_lds[1][128 * 32];
  __shared__ bf16_t b_lds[1][128 * 32];
#endif

  const int lane = threadIdx.x & 31;
  const int wave = threadIdx.x >> 5;
  const int half = lane >> 4;
  const int lr   = lane & 15;
  const int wm   = wave & 3;   // M offset wm*32
  const int wn   = wave >> 2;  // N offset wn*64
  const int mtile = blockIdx.x * 128;
  const int ntile = blockIdx.y * 128;

  v8f acc[2][4] = {};

#if USE_TDM
  const bf16_t* abase = A + (size_t)mtile * DIM_;
  const bf16_t* bbase = W + (size_t)ntile * DIM_;
  if (wave == 0) {
    tdm_load_tile(&a_lds[0][0], abase);
    tdm_load_tile(&b_lds[0][0], bbase);
  }
#else
  const int crow = threadIdx.x >> 1;
  const int cseg = (threadIdx.x & 1) * 16;
  const bf16_t* agp = A + (size_t)(mtile + crow) * DIM_ + cseg;
  const bf16_t* bgp = W + (size_t)(ntile + crow) * DIM_ + cseg;
  bf16_t* alp = &a_lds[0][crow * 32 + cseg];
  bf16_t* blp = &b_lds[0][crow * 32 + cseg];
#endif

  for (int kb = 0; kb < DIM_; kb += 32) {
#if USE_TDM
    const int buf = (kb >> 5) & 1;
    if (wave == 0) __builtin_amdgcn_s_wait_tensorcnt(0);
    __syncthreads();
    if (wave == 0 && kb + 32 < DIM_) {
      tdm_load_tile(&a_lds[buf ^ 1][0], abase + kb + 32);
      tdm_load_tile(&b_lds[buf ^ 1][0], bbase + kb + 32);
    }
    const bf16_t* al = &a_lds[buf][0];
    const bf16_t* bl = &b_lds[buf][0];
#else
    ((v8bf*)alp)[0] = ((const v8bf*)(agp + kb))[0];
    ((v8bf*)alp)[1] = ((const v8bf*)(agp + kb))[1];
    ((v8bf*)blp)[0] = ((const v8bf*)(bgp + kb))[0];
    ((v8bf*)blp)[1] = ((const v8bf*)(bgp + kb))[1];
    __syncthreads();
    const bf16_t* al = &a_lds[0][0];
    const bf16_t* bl = &b_lds[0][0];
#endif

    // A fragments: lane lr = row, K runs {half*8..+7} and {16+half*8..+7}
    v16bf afr[2];
#pragma unroll
    for (int mf = 0; mf < 2; ++mf) {
      const bf16_t* ar = al + (wm * 32 + mf * 16 + lr) * 32;
      v8bf lo = *(const v8bf*)(ar + half * 8);
      v8bf hi = *(const v8bf*)(ar + 16 + half * 8);
      afr[mf] = __builtin_shufflevector(lo, hi, 0,1,2,3,4,5,6,7,8,9,10,11,12,13,14,15);
    }
#pragma unroll
    for (int nt = 0; nt < 4; ++nt) {
      // B fragment: lane lr = col, K runs {half*8..+7} and {16+half*8..+7}
      const bf16_t* br = bl + (wn * 64 + nt * 16 + lr) * 32 + half * 8;
      v8bf lo = *(const v8bf*)br;
      v8bf hi = *(const v8bf*)(br + 16);
      v16bf bfr = __builtin_shufflevector(lo, hi, 0,1,2,3,4,5,6,7,8,9,10,11,12,13,14,15);
      acc[0][nt] = __builtin_amdgcn_wmma_f32_16x16x32_bf16(
          false, afr[0], false, bfr, (short)0, acc[0][nt], false, false);
      acc[1][nt] = __builtin_amdgcn_wmma_f32_16x16x32_bf16(
          false, afr[1], false, bfr, (short)0, acc[1][nt], false, false);
    }
#if !USE_TDM
    __syncthreads();
#endif
  }

  // epilogue
#pragma unroll
  for (int mf = 0; mf < 2; ++mf) {
#pragma unroll
    for (int nt = 0; nt < 4; ++nt) {
#pragma unroll
      for (int e = 0; e < 8; ++e) {
        int gm = mtile + wm * 32 + mf * 16 + e + half * 8;
        int gn = ntile + wn * 64 + nt * 16 + lr;
        float val = acc[mf][nt][e];
        if (MODE == 2) {
          ((float*)dstv)[(size_t)gm * DIM_ + gn] = val;
        } else {
          int d = gn & (HD_ - 1);
          if (MODE == 0 && d < 64) {
            // in-register RoPE: partner d+-32 is acc[mf][nt^2], same lane/e
            float partner = acc[mf][nt ^ 2][e];
            float tf = (float)(gm & (T_ - 1));
            float ang = tf * __expf(-(float)d * 0.14391156516f);  // ln(1e4)/64
            float sv, cv;
            __sincosf(ang, &sv, &cv);
            val = (d < 32) ? (val * cv - partner * sv) : (val * cv + partner * sv);
          }
          int b  = gm >> 11;          // / T_
          int tt = gm & (T_ - 1);
          int h  = gn >> 7;           // / HD_
          ((bf16_t*)dstv)[(((size_t)b * NH_ + h) * T_ + tt) * HD_ + d] = f2bf(val);
        }
      }
    }
  }
}

// ---------------------------------------------------------------------------
// Flash attention: block = one (b,h) x 128 queries; 8 waves x 16 q-rows.
// ---------------------------------------------------------------------------
__global__ __launch_bounds__(256)
void flash_attn_kernel(const bf16_t* __restrict__ Q, const bf16_t* __restrict__ K,
                       const bf16_t* __restrict__ V, bf16_t* __restrict__ Y)
{
  __shared__ bf16_t k_lds[32 * 128];    // [key][d]
  __shared__ bf16_t vt_lds[128 * 32];   // [d][key]
  __shared__ bf16_t p_lds[8 * 16 * 32]; // per-wave [row][key]

  const int lane = threadIdx.x & 31;
  const int wave = threadIdx.x >> 5;
  const int half = lane >> 4;
  const int lr   = lane & 15;
  const int bh   = blockIdx.y;
  const int q0   = blockIdx.x * 128 + wave * 16;
  const size_t base = (size_t)bh * T_ * HD_;

  // preload Q fragments (16 rows x 128 d = 4 A-fragments held in VGPRs)
  v16bf qA[4];
  {
    const bf16_t* qr = Q + base + (size_t)(q0 + lr) * HD_;
#pragma unroll
    for (int db = 0; db < 4; ++db) {
      v8bf lo = *(const v8bf*)(qr + db * 32 + half * 16);
      v8bf hi = *(const v8bf*)(qr + db * 32 + half * 16 + 8);
      qA[db] = __builtin_shufflevector(lo, hi, 0,1,2,3,4,5,6,7,8,9,10,11,12,13,14,15);
    }
  }

  v8f O[8] = {};
  float mi[8], li[8];
#pragma unroll
  for (int e = 0; e < 8; ++e) { mi[e] = -3.0e38f; li[e] = 0.f; }

  const float scale = 0.08838834764831845f;  // 1/sqrt(128)
  bf16_t* pw = &p_lds[wave * 16 * 32];

  for (int kb0 = 0; kb0 < T_; kb0 += 32) {
    { // cooperative staging: K row-major, V transposed
      int tid  = threadIdx.x;
      int row  = tid >> 3;
      int dsrt = (tid & 7) * 16;
      const bf16_t* ks = K + base + (size_t)(kb0 + row) * HD_ + dsrt;
      const bf16_t* vs = V + base + (size_t)(kb0 + row) * HD_ + dsrt;
      ((v8bf*)(k_lds + row * 128 + dsrt))[0] = ((const v8bf*)ks)[0];
      ((v8bf*)(k_lds + row * 128 + dsrt))[1] = ((const v8bf*)ks)[1];
#pragma unroll
      for (int j = 0; j < 16; ++j) vt_lds[(dsrt + j) * 32 + row] = vs[j];
      if (kb0 + 32 < T_) {
        __builtin_prefetch(ks + 32 * HD_, 0, 3);
        __builtin_prefetch(vs + 32 * HD_, 0, 3);
      }
    }
    __syncthreads();

    // S (16 q x 32 keys): two 16x16 accumulators, K=128 -> 4 steps
    v8f S0 = {}, S1 = {};
#pragma unroll
    for (int db = 0; db < 4; ++db) {
      int koff = db * 32 + half * 16;
      const bf16_t* p0 = &k_lds[lr * 128 + koff];
      const bf16_t* p1 = &k_lds[(16 + lr) * 128 + koff];
      v16bf b0 = __builtin_shufflevector(*(const v8bf*)p0, *(const v8bf*)(p0 + 8),
                                         0,1,2,3,4,5,6,7,8,9,10,11,12,13,14,15);
      v16bf b1 = __builtin_shufflevector(*(const v8bf*)p1, *(const v8bf*)(p1 + 8),
                                         0,1,2,3,4,5,6,7,8,9,10,11,12,13,14,15);
      S0 = __builtin_amdgcn_wmma_f32_16x16x32_bf16(false, qA[db], false, b0, (short)0, S0, false, false);
      S1 = __builtin_amdgcn_wmma_f32_16x16x32_bf16(false, qA[db], false, b1, (short)0, S1, false, false);
    }

    // streaming softmax; row r = e + half*8 lives in one VGPR across a
    // 16-lane row -> DPP row_ror reductions
    float p0v[8], p1v[8];
#pragma unroll
    for (int e = 0; e < 8; ++e) {
      float s0 = S0[e] * scale, s1 = S1[e] * scale;
      float rm = row_max16(fmaxf(s0, s1));
      float mn    = fmaxf(mi[e], rm);
      float alpha = __expf(mi[e] - mn);
      float e0 = __expf(s0 - mn), e1 = __expf(s1 - mn);
      float rs = row_sum16(e0 + e1);
      li[e] = li[e] * alpha + rs;
      mi[e] = mn;
#pragma unroll
      for (int dt = 0; dt < 8; ++dt) O[dt][e] *= alpha;
      p0v[e] = e0; p1v[e] = e1;
    }

    // stage P (C layout) -> wave-private LDS -> reload as A fragment
#pragma unroll
    for (int e = 0; e < 8; ++e) {
      int row = e + half * 8;
      pw[row * 32 + lr]      = f2bf(p0v[e]);
      pw[row * 32 + 16 + lr] = f2bf(p1v[e]);
    }
    v16bf pa;
    {
      const bf16_t* pr = pw + lr * 32;
      v8bf lo = *(const v8bf*)(pr + half * 16);
      v8bf hi = *(const v8bf*)(pr + half * 16 + 8);
      pa = __builtin_shufflevector(lo, hi, 0,1,2,3,4,5,6,7,8,9,10,11,12,13,14,15);
    }

    // O += P V  (8 d-tiles of 16)
#pragma unroll
    for (int dt = 0; dt < 8; ++dt) {
      const bf16_t* vr = &vt_lds[(dt * 16 + lr) * 32 + half * 16];
      v16bf vb = __builtin_shufflevector(*(const v8bf*)vr, *(const v8bf*)(vr + 8),
                                         0,1,2,3,4,5,6,7,8,9,10,11,12,13,14,15);
      O[dt] = __builtin_amdgcn_wmma_f32_16x16x32_bf16(false, pa, false, vb, (short)0, O[dt], false, false);
    }
    __syncthreads();
  }

  // normalize; write y back to (b, t, dim) layout, bf16
  int b = bh / NH_, h = bh % NH_;
#pragma unroll
  for (int dt = 0; dt < 8; ++dt) {
#pragma unroll
    for (int e = 0; e < 8; ++e) {
      int m    = q0 + e + half * 8;
      int dcol = dt * 16 + lr;
      float val = O[dt][e] / li[e];
      Y[((size_t)(b * T_) + m) * DIM_ + h * HD_ + dcol] = f2bf(val);
    }
  }
}

// ---------------------------------------------------------------------------
extern "C" void kernel_launch(void* const* d_in, const int* in_sizes, int n_in,
                              void* d_out, int out_size, void* d_ws, size_t ws_size,
                              hipStream_t stream)
{
  const float* x  = (const float*)d_in[0];
  const float* wq = (const float*)d_in[1];
  const float* wk = (const float*)d_in[2];
  const float* wv = (const float*)d_in[3];
  const float* wo = (const float*)d_in[4];
  float* out = (float*)d_out;

  const size_t nx = (size_t)M_ * DIM_;     // 8Mi
  const size_t nw = (size_t)DIM_ * DIM_;   // 4Mi
  bf16_t* xb  = (bf16_t*)d_ws;
  bf16_t* wqb = xb  + nx;
  bf16_t* wkb = wqb + nw;
  bf16_t* wvb = wkb + nw;
  bf16_t* wob = wvb + nw;
  bf16_t* qb  = wob + nw;
  bf16_t* kb  = qb  + nx;
  bf16_t* vb  = kb  + nx;
  bf16_t* yb  = vb  + nx;   // total 56Mi bf16 = 112 MB workspace

  // one-time f32 -> bf16 conversions (memory bound)
  f32_to_bf16_kernel<<<(int)(nx / 2048), 256, 0, stream>>>(x,  xb,  (int)nx);
  f32_to_bf16_kernel<<<(int)(nw / 2048), 256, 0, stream>>>(wq, wqb, (int)nw);
  f32_to_bf16_kernel<<<(int)(nw / 2048), 256, 0, stream>>>(wk, wkb, (int)nw);
  f32_to_bf16_kernel<<<(int)(nw / 2048), 256, 0, stream>>>(wv, wvb, (int)nw);
  f32_to_bf16_kernel<<<(int)(nw / 2048), 256, 0, stream>>>(wo, wob, (int)nw);

  dim3 gproj(M_ / 128, DIM_ / 128);   // 32 x 16
  gemm_nt_kernel<0><<<gproj, 256, 0, stream>>>(xb, wqb, qb);
  gemm_nt_kernel<0><<<gproj, 256, 0, stream>>>(xb, wkb, kb);
  gemm_nt_kernel<1><<<gproj, 256, 0, stream>>>(xb, wvb, vb);

  dim3 gf(T_ / 128, B_ * NH_);        // 16 x 32
  flash_attn_kernel<<<gf, 256, 0, stream>>>(qb, kb, vb, yb);

  gemm_nt_kernel<2><<<gproj, 256, 0, stream>>>(yb, wob, out);
}
